// Block_21457656610847
// MI455X (gfx1250) — matrix operand order repaired and checked by reading, not compiled, not run on hardware
//
#include <hip/hip_runtime.h>
#include <hip/hip_bf16.h>
#include <math.h>

#define B_ 2
#define T_ 2048
#define C_ 1024
#define H_ 16
#define N_ 64
#define BT_ (B_*T_)
#define C4_ (4*C_)

typedef __attribute__((ext_vector_type(16))) __bf16        v16bf;
typedef __attribute__((ext_vector_type(8)))  float         v8f;
typedef __attribute__((ext_vector_type(8)))  unsigned int  v8u;

enum { GEMM_EPI_F32 = 0, GEMM_EPI_ADD_F32 = 1, GEMM_EPI_RELU2_BF16 = 2 };

__device__ __forceinline__ unsigned short f2bf(float f) {
  unsigned u = __float_as_uint(f);
  u += 0x7FFFu + ((u >> 16) & 1u);           // round-to-nearest-even
  return (unsigned short)(u >> 16);
}
__device__ __forceinline__ float bf2f(unsigned short h) {
  return __uint_as_float(((unsigned)h) << 16);
}
__device__ __forceinline__ float sigmoidf_(float x) { return 1.0f / (1.0f + expf(-x)); }

// ---------------------------------------------------------------- transpose + convert: dst[n*K+k] = bf16(src[k*N+n])
__global__ void __launch_bounds__(256) cvtT_kernel(const float* __restrict__ src,
                                                   unsigned short* __restrict__ dst,
                                                   int K, int N) {
  __shared__ float tile[32][33];
  int tx = threadIdx.x & 31, ty = threadIdx.x >> 5;   // ty 0..7
  int k0 = blockIdx.y * 32, n0 = blockIdx.x * 32;
#pragma unroll
  for (int i = 0; i < 4; ++i) {
    int k = k0 + ty + i * 8;
    tile[ty + i * 8][tx] = src[(size_t)k * N + n0 + tx];
  }
  __syncthreads();
#pragma unroll
  for (int i = 0; i < 4; ++i) {
    int n = n0 + ty + i * 8;
    dst[(size_t)n * K + k0 + tx] = f2bf(tile[tx][ty + i * 8]);
  }
}

// ---------------------------------------------------------------- LN1 (per token)
__global__ void __launch_bounds__(256) ln1_kernel(
    const float* __restrict__ x, const float* __restrict__ w, const float* __restrict__ bsh,
    float* __restrict__ xn, float* __restrict__ out_last) {
  int tok = blockIdx.x, tid = threadIdx.x;
  int b = tok / T_, t = tok % T_;
  const float* xrow = x + (size_t)tok * C_;
  int c0 = tid * 4;
  float v[4]; float sum = 0.f, sq = 0.f;
#pragma unroll
  for (int m = 0; m < 4; ++m) { v[m] = xrow[c0 + m]; sum += v[m]; sq += v[m] * v[m]; }
  __shared__ float s1[256], s2[256];
  s1[tid] = sum; s2[tid] = sq; __syncthreads();
  for (int s = 128; s > 0; s >>= 1) {
    if (tid < s) { s1[tid] += s1[tid + s]; s2[tid] += s2[tid + s]; }
    __syncthreads();
  }
  float mean = s1[0] * (1.0f / C_);
  float var  = s2[0] * (1.0f / C_) - mean * mean;
  float inv  = rsqrtf(var + 1e-5f);
  float* xo = xn + (size_t)tok * C_;
#pragma unroll
  for (int m = 0; m < 4; ++m) {
    int c = c0 + m;
    float y = (v[m] - mean) * inv * w[c] + bsh[c];
    xo[c] = y;
    if (t == T_ - 1) out_last[(size_t)b * C_ + c] = y;
  }
}

// ---------------------------------------------------------------- token shift + 6 mixes
__global__ void mix6_kernel(const float* __restrict__ xn, const float* __restrict__ att_shift,
                            const float* __restrict__ pr, const float* __restrict__ pw,
                            const float* __restrict__ pk, const float* __restrict__ pv,
                            const float* __restrict__ pa, const float* __restrict__ pg,
                            unsigned short* __restrict__ oxr, unsigned short* __restrict__ oxw,
                            unsigned short* __restrict__ oxk, unsigned short* __restrict__ oxv,
                            unsigned short* __restrict__ oxa, unsigned short* __restrict__ oxg) {
  size_t idx = (size_t)blockIdx.x * blockDim.x + threadIdx.x;
  if (idx >= (size_t)BT_ * C_) return;
  int c = (int)(idx & (C_ - 1));
  size_t tok = idx >> 10;
  int t = (int)(tok % T_), b = (int)(tok / T_);
  float v = xn[idx];
  float prev = (t == 0) ? att_shift[(size_t)b * C_ + c] : xn[idx - C_];
  float xx = prev - v;
  oxr[idx] = f2bf(v + xx * pr[c]);
  oxw[idx] = f2bf(v + xx * pw[c]);
  oxk[idx] = f2bf(v + xx * pk[c]);
  oxv[idx] = f2bf(v + xx * pv[c]);
  oxa[idx] = f2bf(v + xx * pa[c]);
  oxg[idx] = f2bf(v + xx * pg[c]);
}

// ---------------------------------------------------------------- fused low-rank proj: act2(bias + act1(x@W1)@W2)
__global__ void __launch_bounds__(256) proj2_kernel(
    const unsigned short* __restrict__ in_bf, const float* __restrict__ W1,
    const float* __restrict__ W2, const float* __restrict__ bias,
    float* __restrict__ outp, int Dh, int act1, int act2) {
  int tok = blockIdx.x, tid = threadIdx.x;
  __shared__ float sx[C_];
  __shared__ float sh[192];
  const unsigned short* xrow = in_bf + (size_t)tok * C_;
  for (int c = tid; c < C_; c += 256) sx[c] = bf2f(xrow[c]);
  __syncthreads();
  for (int d = tid; d < Dh; d += 256) {
    float acc = 0.f;
    for (int c = 0; c < C_; ++c) acc += sx[c] * W1[(size_t)c * Dh + d];
    if (act1 == 1) acc = tanhf(acc);
    else if (act1 == 2) acc = sigmoidf_(acc);
    sh[d] = acc;
  }
  __syncthreads();
  float* op = outp + (size_t)tok * C_;
  for (int c = tid; c < C_; c += 256) {
    float acc = bias ? bias[c] : 0.0f;
    for (int d = 0; d < Dh; ++d) acc += sh[d] * W2[(size_t)d * C_ + c];
    if (act2 == 2) acc = sigmoidf_(acc);
    op[c] = acc;
  }
}

// ---------------------------------------------------------------- WMMA bf16 GEMM: C = A[M,K] @ B[K,N], B given transposed [N,K]
// LDS tiles are stored in fragment-ready (swizzled) layout so that every
// fragment load is two ds_load_b128 per lane (no scalar gathers).
__global__ void __launch_bounds__(256) gemm_bf16_wmma_kernel(
    const unsigned short* __restrict__ A,   // [M,K] row-major bf16
    const unsigned short* __restrict__ Bt,  // [N,K] row-major bf16 (pre-transposed weights)
    float* __restrict__ Cf, unsigned short* __restrict__ Cbf,
    const float* __restrict__ addp, int M, int Nn, int K, int epi) {
  // sA[sub][lane][8 dwords]: A fragment layout, sub = 16-row group of the 128-row tile
  // sB[sub][lane][8 dwords]: B fragment layout, sub = 16-col group of the 128-col tile
  __shared__ alignas(32) unsigned int sA[8][32][8];
  __shared__ alignas(32) unsigned int sB[8][32][8];
  int tid = threadIdx.x;
  int lane = tid & 31, wave = tid >> 5;
  int mBase = (wave >> 1) * 32;                 // 4 waves stacked in M
  int nBase = (wave & 1) * 64;                  // 2 waves tiled in N
  int rowBlock = blockIdx.y * 128;
  int colBlock = blockIdx.x * 128;
  int lrow = tid >> 1;                          // 0..127 (tile row for A, tile col for B)
  int lch  = tid & 1;                           // which 16-wide k chunk
  int asub = lrow >> 4, am = lrow & 15;
  int bsub = lrow >> 4, bl = lrow & 15;
  int half = lane >> 4, ncol = lane & 15;

  v8f acc[2][4];
  for (int i = 0; i < 2; ++i)
    for (int j = 0; j < 4; ++j)
      for (int z = 0; z < 8; ++z) acc[i][j][z] = 0.0f;

  int nK = K >> 5;
  for (int kb = 0; kb < nK; ++kb) {
    {  // A tile 128x32 -> fragment-ready LDS
      const uint4* ga = (const uint4*)(A + (size_t)(rowBlock + lrow) * K + kb * 32 + lch * 16);
      uint4 a0 = ga[0], a1 = ga[1];               // k-chunk 0..7 / 8..15 of this 16-k group
      *(uint4*)&sA[asub][am][lch * 4]      = a0;  // lanes 0-15 half
      *(uint4*)&sA[asub][am + 16][lch * 4] = a1;  // lanes 16-31 half
      if (kb + 1 < nK) __builtin_prefetch((const unsigned short*)ga + 32, 0, 1);
    }
    {  // B tile (transposed): 128 cols x 32 k -> fragment-ready LDS
      const uint4* gb = (const uint4*)(Bt + (size_t)(colBlock + lrow) * K + kb * 32 + lch * 16);
      uint4 b0 = gb[0], b1 = gb[1];
      *(uint4*)&sB[bsub][bl][lch * 4]      = b0;  // lanes 0-15 half
      *(uint4*)&sB[bsub][bl + 16][lch * 4] = b1;  // lanes 16-31 half
      if (kb + 1 < nK) __builtin_prefetch((const unsigned short*)gb + 32, 0, 1);
    }
    __syncthreads();

    v16bf afrag[2];
#pragma unroll
    for (int i = 0; i < 2; ++i) {
      v8u raw = *(const v8u*)&sA[(mBase >> 4) + i][lane][0];
      afrag[i] = __builtin_bit_cast(v16bf, raw);
    }
    v16bf bfrag[4];
#pragma unroll
    for (int j = 0; j < 4; ++j) {
      v8u raw = *(const v8u*)&sB[(nBase >> 4) + j][lane][0];
      bfrag[j] = __builtin_bit_cast(v16bf, raw);
    }
#pragma unroll
    for (int i = 0; i < 2; ++i)
#pragma unroll
      for (int j = 0; j < 4; ++j)
        acc[i][j] = __builtin_amdgcn_wmma_f32_16x16x32_bf16(
            false, afrag[i], false, bfrag[j], (short)0, acc[i][j], false, false);
    __syncthreads();
  }

  // epilogue (C/D layout: VGPR p -> M=p (lanes<16) / p+8 (lanes>=16), N=lane&15)
#pragma unroll
  for (int i = 0; i < 2; ++i) {
#pragma unroll
    for (int j = 0; j < 4; ++j) {
      int gcol = colBlock + nBase + j * 16 + ncol;
#pragma unroll
      for (int p = 0; p < 8; ++p) {
        int grow = rowBlock + mBase + i * 16 + p + 8 * half;
        size_t idx = (size_t)grow * Nn + gcol;
        float val = acc[i][j][p];
        if (epi == GEMM_EPI_F32)           Cf[idx] = val;
        else if (epi == GEMM_EPI_ADD_F32)  Cf[idx] = val + addp[idx];
        else { float r = fmaxf(val, 0.0f); Cbf[idx] = f2bf(r * r); }
      }
    }
  }
}

// ---------------------------------------------------------------- pre-scan elementwise (per token, per-head reductions)
__global__ void __launch_bounds__(256) prescan_kernel(
    float* __restrict__ kbuf, const float* __restrict__ amix, float* __restrict__ wwbuf,
    float* __restrict__ vbuf, const float* __restrict__ mv, const float* __restrict__ v_first,
    float* __restrict__ aabuf, float* __restrict__ bbbuf,
    const float* __restrict__ k_k, const float* __restrict__ k_a) {
  int tok = blockIdx.x, tid = threadIdx.x;
  int h = tid >> 4, q = tid & 15;
  int cb = h * 64 + q * 4;
  size_t base = (size_t)tok * C_ + cb;
  float kv[4], av[4], kkv[4];
  float ss = 0.f;
#pragma unroll
  for (int m = 0; m < 4; ++m) {
    kv[m] = kbuf[base + m];
    av[m] = amix[base + m];
    kkv[m] = kv[m] * k_k[cb + m];
    ss += kkv[m] * kkv[m];
  }
  __shared__ float red[16][16];
  __shared__ float hinv[16];
  red[h][q] = ss; __syncthreads();
  if (q == 0) {
    float s = 0.f;
    for (int e = 0; e < 16; ++e) s += red[h][e];
    hinv[h] = 1.0f / fmaxf(sqrtf(s), 1e-12f);
  }
  __syncthreads();
  float iv = hinv[h];
#pragma unroll
  for (int m = 0; m < 4; ++m) {
    float kkn = kkv[m] * iv;
    aabuf[base + m] = -kkn;                                   // a_t = -kk
    bbbuf[base + m] = kkn * av[m];                            // b_t = kk*a
    kbuf[base + m]  = kv[m] * (1.f + (av[m] - 1.f) * k_a[cb + m]);  // k2
    float ww = wwbuf[base + m];
    float wl = -log1pf(expf(-ww)) - 0.5f;                     // -softplus(-ww) - 0.5
    wwbuf[base + m] = expf(-expf(wl));                        // decay
    float vv = vbuf[base + m];
    vbuf[base + m] = vv + (v_first[base + m] - vv) * mv[base + m];
  }
}

// ---------------------------------------------------------------- sequential RWKV7 scan (one block per (b,h))
__global__ void __launch_bounds__(64) scan_kernel(
    const float* __restrict__ decay, const float* __restrict__ rb,
    const float* __restrict__ kb, const float* __restrict__ vb,
    const float* __restrict__ aab, const float* __restrict__ bbb,
    const float* __restrict__ S0, float* __restrict__ ob, float* __restrict__ Sout) {
  int bh = blockIdx.x;
  int b = bh / H_, h = bh % H_;
  int i = threadIdx.x;                           // state row (value dim)
  float S[64];
  {
    const float* s0 = S0 + (size_t)bh * N_ * N_ + (size_t)i * N_;
#pragma unroll
    for (int j = 0; j < 64; ++j) S[j] = s0[j];
  }
  __shared__ float sd[64], sr[64], sk[64], sv[64], sa[64], sb[64];
  size_t stride = (size_t)H_ * N_;
  size_t base = ((size_t)b * T_ * H_ + h) * N_ + i;
  for (int t = 0; t < T_; ++t) {
    size_t p = base + (size_t)t * stride;
    sd[i] = decay[p]; sr[i] = rb[p]; sk[i] = kb[p];
    sv[i] = vb[p];    sa[i] = aab[p]; sb[i] = bbb[p];
    __syncthreads();
    float sacc = 0.f;
#pragma unroll 8
    for (int j = 0; j < 64; ++j) sacc += S[j] * sa[j];
    float vi = sv[i], oacc = 0.f;
#pragma unroll 8
    for (int j = 0; j < 64; ++j) {
      float s2 = S[j] * sd[j] + sacc * sb[j] + vi * sk[j];
      S[j] = s2;
      oacc += s2 * sr[j];
    }
    ob[p] = oacc;
    __syncthreads();
  }
  float* so = Sout + (size_t)bh * N_ * N_ + (size_t)i * N_;
#pragma unroll
  for (int j = 0; j < 64; ++j) so[j] = S[j];
}

// ---------------------------------------------------------------- GroupNorm + residual + gate -> bf16
__global__ void __launch_bounds__(256) postscan_kernel(
    const float* __restrict__ o, const float* __restrict__ rb, const float* __restrict__ k2,
    const float* __restrict__ vb, const float* __restrict__ gb, const float* __restrict__ r_k,
    const float* __restrict__ gn_w, const float* __restrict__ gn_b,
    unsigned short* __restrict__ ybf) {
  int tok = blockIdx.x, tid = threadIdx.x;
  int h = tid >> 4, q = tid & 15;
  int cb = h * 64 + q * 4;
  size_t base = (size_t)tok * C_ + cb;
  float ov[4];
  float sum = 0.f, sq = 0.f, dot = 0.f;
#pragma unroll
  for (int m = 0; m < 4; ++m) {
    ov[m] = o[base + m];
    sum += ov[m]; sq += ov[m] * ov[m];
    dot += rb[base + m] * k2[base + m] * r_k[cb + m];
  }
  __shared__ float red[16][16];
  __shared__ float hsum[16], hsq[16], hdot[16];
  red[h][q] = sum; __syncthreads();
  if (q == 0) { float s = 0.f; for (int e = 0; e < 16; ++e) s += red[h][e]; hsum[h] = s; } __syncthreads();
  red[h][q] = sq; __syncthreads();
  if (q == 0) { float s = 0.f; for (int e = 0; e < 16; ++e) s += red[h][e]; hsq[h] = s; } __syncthreads();
  red[h][q] = dot; __syncthreads();
  if (q == 0) { float s = 0.f; for (int e = 0; e < 16; ++e) s += red[h][e]; hdot[h] = s; } __syncthreads();
  float mean = hsum[h] * (1.0f / N_);
  float var  = hsq[h] * (1.0f / N_) - mean * mean;
  float inv  = rsqrtf(var + 6.4e-4f);            // 1e-5 * 8^2
#pragma unroll
  for (int m = 0; m < 4; ++m) {
    float on = (ov[m] - mean) * inv * gn_w[cb + m] + gn_b[cb + m];
    float y = (on + hdot[h] * vb[base + m]) * gb[base + m];
    ybf[base + m] = f2bf(y);
  }
}

// ---------------------------------------------------------------- LN2 + ffn mix -> bf16 (+ last-token xn2 out)
__global__ void __launch_bounds__(256) ln2_mix_kernel(
    const float* __restrict__ x1, const float* __restrict__ w, const float* __restrict__ bsh,
    const float* __restrict__ ffn_shift, const float* __restrict__ fx_k,
    unsigned short* __restrict__ kf_bf, float* __restrict__ out_last) {
  int tok = blockIdx.x, tid = threadIdx.x;
  int b = tok / T_, t = tok % T_;
  const float* xrow = x1 + (size_t)tok * C_;
  int c0 = tid * 4;
  float v[4]; float sum = 0.f, sq = 0.f;
#pragma unroll
  for (int m = 0; m < 4; ++m) { v[m] = xrow[c0 + m]; sum += v[m]; sq += v[m] * v[m]; }
  __shared__ float s1[256], s2[256];
  s1[tid] = sum; s2[tid] = sq; __syncthreads();
  for (int s = 128; s > 0; s >>= 1) {
    if (tid < s) { s1[tid] += s1[tid + s]; s2[tid] += s2[tid + s]; }
    __syncthreads();
  }
  float mean = s1[0] * (1.0f / C_);
  float var  = s2[0] * (1.0f / C_) - mean * mean;
  float inv  = rsqrtf(var + 1e-5f);
#pragma unroll
  for (int m = 0; m < 4; ++m) {
    int c = c0 + m;
    float xn2 = (v[m] - mean) * inv * w[c] + bsh[c];
    if (t == T_ - 1) out_last[(size_t)b * C_ + c] = xn2;
    float kf = xn2 + (ffn_shift[(size_t)b * C_ + c] - xn2) * fx_k[c];
    kf_bf[(size_t)tok * C_ + c] = f2bf(kf);
  }
}

// ================================================================ host
extern "C" void kernel_launch(void* const* d_in, const int* in_sizes, int n_in,
                              void* d_out, int out_size, void* d_ws, size_t ws_size,
                              hipStream_t stream) {
  enum {
    IN_X = 0, IN_VFIRST, IN_ATT, IN_WKV, IN_FFN,
    P_XR, P_XW, P_XK, P_XV, P_XA, P_XG,
    P_WR, P_WK, P_WV, P_WO,
    P_W0, P_W1, P_W2, P_A0, P_A1, P_A2, P_V0, P_V1, P_V2,
    P_G1, P_G2, P_KK, P_KA, P_RK, P_GNW, P_GNB,
    P_LN1W, P_LN1B, P_LN2W, P_LN2B, P_FXK, P_FWK, P_FWV
  };
  (void)in_sizes; (void)n_in; (void)out_size;

  const float* x        = (const float*)d_in[IN_X];
  const float* v_first  = (const float*)d_in[IN_VFIRST];
  const float* att_sh   = (const float*)d_in[IN_ATT];
  const float* wkv0     = (const float*)d_in[IN_WKV];
  const float* ffn_sh   = (const float*)d_in[IN_FFN];

  // d_out regions
  float* out       = (float*)d_out;
  const size_t BTC = (size_t)BT_ * C_;
  float* out_x2      = out;
  float* out_vfirst  = out + BTC;
  float* out_xnlast  = out + 2 * BTC;
  float* out_S       = out_xnlast + (size_t)B_ * C_;
  float* out_xn2last = out_S + (size_t)B_ * H_ * N_ * N_;

  // workspace carve-out
  char* ws = (char*)d_ws;
  size_t off = 0;
  auto alloc = [&](size_t bytes) -> char* {
    char* p = ws + off;
    off = (off + bytes + 255) & ~(size_t)255;
    return p;
  };
  const size_t FB = BTC * sizeof(float);
  const size_t HB = BTC * sizeof(unsigned short);

  float* xn            = (float*)alloc(FB);
  unsigned short* xrB  = (unsigned short*)alloc(HB);
  unsigned short* xwB  = (unsigned short*)alloc(HB);
  unsigned short* xkB  = (unsigned short*)alloc(HB);
  unsigned short* xvB  = (unsigned short*)alloc(HB);
  unsigned short* xaB  = (unsigned short*)alloc(HB);
  unsigned short* xgB  = (unsigned short*)alloc(HB);
  unsigned short* WrT  = (unsigned short*)alloc((size_t)C_ * C_ * 2);   // [N,K] bf16
  unsigned short* WkT  = (unsigned short*)alloc((size_t)C_ * C_ * 2);
  unsigned short* WvT  = (unsigned short*)alloc((size_t)C_ * C_ * 2);
  unsigned short* WoT  = (unsigned short*)alloc((size_t)C_ * C_ * 2);
  unsigned short* fWkT = (unsigned short*)alloc((size_t)C_ * C4_ * 2);  // [4C, C]
  unsigned short* fWvT = (unsigned short*)alloc((size_t)C4_ * C_ * 2);  // [C, 4C]
  float* rbuf  = (float*)alloc(FB);
  float* kbuf  = (float*)alloc(FB);   // becomes k2 in-place
  float* vbuf  = (float*)alloc(FB);   // becomes mixed v in-place
  float* wwbuf = (float*)alloc(FB);   // w0 + tanh(xw@w1)@w2, then decay in-place
  float* amix  = (float*)alloc(FB);
  float* mvbuf = (float*)alloc(FB);
  float* gbuf  = (float*)alloc(FB);
  float* aabuf = (float*)alloc(FB);
  float* bbbuf = (float*)alloc(FB);
  float* obuf  = (float*)alloc(FB);
  unsigned short* ybf  = (unsigned short*)alloc(HB);
  float* x1buf = (float*)alloc(FB);
  unsigned short* kfB  = (unsigned short*)alloc(HB);
  unsigned short* kfhB = (unsigned short*)alloc((size_t)BT_ * C4_ * 2);
  (void)ws_size;

  // weight transpose+convert: src [K,N] f32 -> dst [N,K] bf16
  auto cvtT = [&](const void* s, unsigned short* d, int K, int N) {
    dim3 g(N / 32, K / 32);
    cvtT_kernel<<<g, 256, 0, stream>>>((const float*)s, d, K, N);
  };
  cvtT(d_in[P_WR], WrT, C_, C_);
  cvtT(d_in[P_WK], WkT, C_, C_);
  cvtT(d_in[P_WV], WvT, C_, C_);
  cvtT(d_in[P_WO], WoT, C_, C_);
  cvtT(d_in[P_FWK], fWkT, C_, C4_);
  cvtT(d_in[P_FWV], fWvT, C4_, C_);

  // 1) LN1 (+ last-token xn output)
  ln1_kernel<<<BT_, 256, 0, stream>>>(x, (const float*)d_in[P_LN1W], (const float*)d_in[P_LN1B],
                                      xn, out_xnlast);
  // 2) token shift + mixes
  mix6_kernel<<<(unsigned)(BTC / 256), 256, 0, stream>>>(
      xn, att_sh,
      (const float*)d_in[P_XR], (const float*)d_in[P_XW], (const float*)d_in[P_XK],
      (const float*)d_in[P_XV], (const float*)d_in[P_XA], (const float*)d_in[P_XG],
      xrB, xwB, xkB, xvB, xaB, xgB);

  // 3) small low-rank projections
  proj2_kernel<<<BT_, 256, 0, stream>>>(xwB, (const float*)d_in[P_W1], (const float*)d_in[P_W2],
                                        (const float*)d_in[P_W0], wwbuf, 64, 1, 0);   // tanh, +w0
  proj2_kernel<<<BT_, 256, 0, stream>>>(xaB, (const float*)d_in[P_A1], (const float*)d_in[P_A2],
                                        (const float*)d_in[P_A0], amix, 64, 0, 2);    // sigmoid(a0+..)
  proj2_kernel<<<BT_, 256, 0, stream>>>(xvB, (const float*)d_in[P_V1], (const float*)d_in[P_V2],
                                        (const float*)d_in[P_V0], mvbuf, 32, 0, 2);   // sigmoid(v0+..)
  proj2_kernel<<<BT_, 256, 0, stream>>>(xgB, (const float*)d_in[P_G1], (const float*)d_in[P_G2],
                                        nullptr, gbuf, 160, 2, 0);                    // sigmoid(.)@g2

  // 4) big GEMMs r/k/v (WMMA)
  dim3 gCC(C_ / 128, BT_ / 128);
  gemm_bf16_wmma_kernel<<<gCC, 256, 0, stream>>>(xrB, WrT, rbuf, nullptr, nullptr, BT_, C_, C_, GEMM_EPI_F32);
  gemm_bf16_wmma_kernel<<<gCC, 256, 0, stream>>>(xkB, WkT, kbuf, nullptr, nullptr, BT_, C_, C_, GEMM_EPI_F32);
  gemm_bf16_wmma_kernel<<<gCC, 256, 0, stream>>>(xvB, WvT, vbuf, nullptr, nullptr, BT_, C_, C_, GEMM_EPI_F32);

  // 5) pre-scan elementwise (decay, kk-norm, k2, value-residual mix)
  prescan_kernel<<<BT_, 256, 0, stream>>>(kbuf, amix, wwbuf, vbuf, mvbuf, v_first,
                                          aabuf, bbbuf,
                                          (const float*)d_in[P_KK], (const float*)d_in[P_KA]);

  // 6) sequential scan -> o, final state S to d_out
  scan_kernel<<<B_ * H_, 64, 0, stream>>>(wwbuf, rbuf, kbuf, vbuf, aabuf, bbbuf, wkv0, obuf, out_S);

  // 7) GroupNorm + residual + gate, then x1 = x + y@Wo
  postscan_kernel<<<BT_, 256, 0, stream>>>(obuf, rbuf, kbuf, vbuf, gbuf,
                                           (const float*)d_in[P_RK],
                                           (const float*)d_in[P_GNW], (const float*)d_in[P_GNB], ybf);
  gemm_bf16_wmma_kernel<<<gCC, 256, 0, stream>>>(ybf, WoT, x1buf, nullptr, x, BT_, C_, C_, GEMM_EPI_ADD_F32);

  // 8) channel mix
  ln2_mix_kernel<<<BT_, 256, 0, stream>>>(x1buf, (const float*)d_in[P_LN2W], (const float*)d_in[P_LN2B],
                                          ffn_sh, (const float*)d_in[P_FXK], kfB, out_xn2last);
  dim3 gC4(C4_ / 128, BT_ / 128);
  gemm_bf16_wmma_kernel<<<gC4, 256, 0, stream>>>(kfB, fWkT, nullptr, kfhB, nullptr, BT_, C4_, C_, GEMM_EPI_RELU2_BF16);
  gemm_bf16_wmma_kernel<<<gCC, 256, 0, stream>>>(kfhB, fWvT, out_x2, nullptr, x1buf, BT_, C_, C4_, GEMM_EPI_ADD_F32);

  // 9) v_first passthrough
  hipMemcpyAsync(out_vfirst, v_first, FB, hipMemcpyDeviceToDevice, stream);
}
